// MultiScaleFeatureExtractor_627065225356
// MI455X (gfx1250) — matrix operand order repaired and checked by reading, not compiled, not run on hardware
//
#include <hip/hip_runtime.h>
#include <hip/hip_bf16.h>
#include <cstdint>
#include <cstddef>

// ---------------------------------------------------------------------------
// MultiScaleFeatureExtractor for gfx1250 (MI455X).
// All 1x1 convs + per-neighbor MLPs mapped to v_wmma_f32_16x16x32_f16.
// BN folded into weights at prep time; activations stored point-major f16.
// GEMMs fully templated (compile-time K-loop, strides, immediate-offset
// loads, A-fragment register reuse across 4 point tiles, bias seeded as the
// WMMA C operand). point3dconv fully fused in registers with __shfl_xor
// D->B fragment chaining; delta bias folded into a weight column.
// ---------------------------------------------------------------------------

typedef __attribute__((ext_vector_type(16))) _Float16 v16h;
typedef __attribute__((ext_vector_type(8)))  _Float16 v8h;
typedef __attribute__((ext_vector_type(8)))  float    v8f;
typedef _Float16 h16;

#define NB   4
#define NPTS 4096
#define NP   (NB * NPTS)     // 16384 points total
#define NFEAT 128
#define PTILES (NP / 16)     // 1024 point tiles

__device__ __forceinline__ v8f wmma16(v16h a, v16h b, v8f c) {
  // D = A(16x32,f16) * B(32x16,f16) + C(16x16,f32)
  return __builtin_amdgcn_wmma_f32_16x16x32_f16(false, a, false, b,
                                                (short)0, c, false, false);
}

// ---------------------------------------------------------------------------
// Fold BN into conv weights, write f16 weights pre-swizzled into the WMMA
// A-fragment per-lane layout: tile (mo,kt) -> 32 lanes x 16 halfs contiguous.
// A k order used here (B loads match): lane<16: row M=lane,    halfs = K{0..15}
//                        lane>=16: row M=lane-16, halfs = K{16..31}
// If bias_in_col >= 0 the folded bias is stored at that (padding) K column,
// to be activated by a 1.0 input channel.
// ---------------------------------------------------------------------------
__global__ void prep_conv(const float* __restrict__ W, const float* __restrict__ b,
                          const float* __restrict__ be, const float* __restrict__ g,
                          int Cin, int Cout, int bias_in_col,
                          h16* __restrict__ wdst, float* __restrict__ bdst) {
  const int KT = (Cin + 31) / 32;
  const int MT = Cout / 16;
  const int total = MT * KT * 512;
  const int tid = blockIdx.x * blockDim.x + threadIdx.x;
  const float rs = rsqrtf(1.0f + 1e-5f);
  if (tid < total) {
    int t = tid >> 9;          // tile index
    int l = (tid >> 4) & 31;   // lane
    int h = tid & 15;          // half slot within lane
    int mo = t / KT, kt = t % KT;
    int m = mo * 16 + (l & 15);
    int kmap = (l < 16) ? h
                        : h + 16;
    int k = kt * 32 + kmap;
    float w;
    if (k < Cin)                w = W[m * Cin + k] * g[m] * rs;
    else if (k == bias_in_col)  w = b[m] * g[m] * rs + be[m];
    else                        w = 0.0f;
    wdst[tid] = (h16)w;
  }
  if (tid < Cout) bdst[tid] = b[tid] * g[tid] * rs + be[tid];
}

// pts (B,3,N) f32 -> point-major f16 rows of 32 (xyz + zero pad)
__global__ void pack_pts(const float* __restrict__ pts, h16* __restrict__ dst) {
  int p = blockIdx.x * blockDim.x + threadIdx.x;
  if (p >= NP) return;
  int b = p / NPTS, n = p % NPTS;
  h16* row = dst + (size_t)p * 32;
  row[0] = (h16)pts[(b * 3 + 0) * NPTS + n];
  row[1] = (h16)pts[(b * 3 + 1) * NPTS + n];
  row[2] = (h16)pts[(b * 3 + 2) * NPTS + n];
  #pragma unroll
  for (int c = 3; c < 32; ++c) row[c] = (h16)0.0f;
}

// ---------------------------------------------------------------------------
// Generic fused GEMM + bias + ReLU over all B*N points.
// in = concat(in0[ kt<KSPLIT ], in1[ kt>=KSPLIT ]) along channels. KT/KSPLIT/
// strides all compile-time: the K-loop unrolls into immediate-offset b128
// loads; p*LD becomes shift/add. Bias is seeded as the WMMA C accumulator.
// Each wave: one 16-outch tile x PPW point tiles with A-fragments in regs.
// out: f16 point-major (LDO halfs/row @ coloff) or f32 channel-major (B,128,N).
// Exact tiling -> EXEC all ones around every WMMA.
// ---------------------------------------------------------------------------
template<int KT, int KSPLIT, int PPW, bool F32OUT, int LD0, int LD1, int LDO>
__global__ void gemm_cbr(const h16* __restrict__ in0,
                         const h16* __restrict__ in1,
                         const h16* __restrict__ Wt,
                         const float* __restrict__ bias,
                         h16* __restrict__ out16, int coloff,
                         float* __restrict__ out32) {
  const int gw   = (blockIdx.x * blockDim.x + threadIdx.x) >> 5;
  const int lane = threadIdx.x & 31;
  const int hi   = lane >> 4;
  const int ntg  = PTILES / PPW;
  const int tg   = gw % ntg;
  const int mo   = gw / ntg;

  // preload A fragments (weights) for all kt
  v16h af[KT];
  #pragma unroll
  for (int kt = 0; kt < KT; ++kt)
    af[kt] = *(const v16h*)(Wt + ((size_t)(mo * KT + kt) * 32 + lane) * 16);

  const int chb = mo * 16 + hi * 8;
  v8f bseed;
  #pragma unroll
  for (int r = 0; r < 8; ++r) bseed[r] = bias[chb + r];

  #pragma unroll
  for (int t = 0; t < PPW; ++t) {
    const int p = (tg * PPW + t) * 16 + (lane & 15);
    const h16* base0 = in0 + (size_t)p * LD0 + hi * 16;
    const h16* base1 = in1 + (size_t)p * LD1 + hi * 16;
    v8f acc = bseed;                 // bias as WMMA C operand
    #pragma unroll
    for (int kt = 0; kt < KT; ++kt) {
      const h16* src = (kt < KSPLIT) ? (base0 + kt * 32)
                                     : (base1 + (kt - KSPLIT) * 32);
      v16h bf = *(const v16h*)src;
      acc = wmma16(af[kt], bf, acc);
    }
    #pragma unroll
    for (int r = 0; r < 8; ++r) acc[r] = acc[r] > 0.0f ? acc[r] : 0.0f;
    if (F32OUT) {
      int b = p / NPTS, n = p % NPTS;
      #pragma unroll
      for (int r = 0; r < 8; ++r)
        out32[((size_t)(b * NFEAT + chb + r)) * NPTS + n] = acc[r];
    } else {
      v8h o;
      #pragma unroll
      for (int r = 0; r < 8; ++r) o[r] = (h16)acc[r];
      *(v8h*)(out16 + (size_t)p * LDO + coloff + chb) = o;
    }
  }
}

// ---------------------------------------------------------------------------
// Fused point3dconv: per (point n, neighbor k):
//   delta = ReLU(Wd*[pts[j]-pts[n];1])     (3->32, bias folded in col 3)
//   f     = ReLU(Wf*bott[j] + bf)          (64->32, gathered; bias as C)
//   q     = ReLU(Wp*(delta*f) + bp)        (32->32; bias as C)
//   new[n] = sum_k q                       -> feats[:, coloff..coloff+32)
// One wave handles P points (P=2 for K=8, else 1), looping S k-subtiles
// (S=2 for K=32). D->B fragment chaining done with __shfl_xor(.,16).
// ---------------------------------------------------------------------------
template<int K>
__global__ void p3dconv(const float* __restrict__ pts, const int* __restrict__ idx,
                        const h16* __restrict__ bott,
                        const h16* __restrict__ Wd,
                        const h16* __restrict__ Wf, const float* __restrict__ bfv,
                        const h16* __restrict__ Wp, const float* __restrict__ bp,
                        h16* __restrict__ feats, int coloff) {
  constexpr int P = (K == 8) ? 2 : 1;
  constexpr int S = (K == 32) ? 2 : 1;
  const int gw   = (blockIdx.x * blockDim.x + threadIdx.x) >> 5;
  const int lane = threadIdx.x & 31;
  const int hi   = lane >> 4;
  const int c    = lane & 15;
  const int pb   = gw * P;             // global point base
  const int b    = pb / NPTS;

  // A fragments (weights), loaded once per wave
  v16h Ad0  = *(const v16h*)(Wd + lane * 16);
  v16h Ad1  = *(const v16h*)(Wd + 512 + lane * 16);
  v16h Af00 = *(const v16h*)(Wf + 0 * 512 + lane * 16);
  v16h Af01 = *(const v16h*)(Wf + 1 * 512 + lane * 16);
  v16h Af10 = *(const v16h*)(Wf + 2 * 512 + lane * 16);
  v16h Af11 = *(const v16h*)(Wf + 3 * 512 + lane * 16);
  v16h Ap0  = *(const v16h*)(Wp + lane * 16);
  v16h Ap1  = *(const v16h*)(Wp + 512 + lane * 16);

  // hoisted bias vectors, used as WMMA C seeds (feats + post convs)
  v8f bfs0, bfs1, bps0, bps1;
  #pragma unroll
  for (int r = 0; r < 8; ++r) {
    bfs0[r] = bfv[hi * 8 + r];
    bfs1[r] = bfv[16 + hi * 8 + r];
    bps0[r] = bp[hi * 8 + r];
    bps1[r] = bp[16 + hi * 8 + r];
  }

  v8f kacc0 = {}, kacc1 = {};

  #pragma unroll
  for (int st = 0; st < S; ++st) {
    int l  = st * 16 + c;
    int n  = pb + l / K;               // global point of this column
    int nb = n - b * NPTS;             // within-batch point
    int kk = l % K;
    int j  = idx[((size_t)b * NPTS + nb) * K + kk];   // neighbor (within batch)

    // ---- delta branch: 3 -> 32 (bias via constant-1 channel) ----
    float dx = pts[(b * 3 + 0) * NPTS + j] - pts[(b * 3 + 0) * NPTS + nb];
    float dy = pts[(b * 3 + 1) * NPTS + j] - pts[(b * 3 + 1) * NPTS + nb];
    float dz = pts[(b * 3 + 2) * NPTS + j] - pts[(b * 3 + 2) * NPTS + nb];
    v16h bdlt = {};
    if (!hi) {
      bdlt[0] = (h16)dx; bdlt[1] = (h16)dy; bdlt[2] = (h16)dz;
      bdlt[3] = (h16)1.0f;
    }
    v8f dd0 = {}, dd1 = {};
    dd0 = wmma16(Ad0, bdlt, dd0);
    dd1 = wmma16(Ad1, bdlt, dd1);

    // ---- feats branch: gather 64 -> 32, bias seeded as C ----
    size_t jrow = ((size_t)b * NPTS + j) * 64;
    v16h bf0 = *(const v16h*)(bott + jrow + hi * 16);
    v16h bf1 = *(const v16h*)(bott + jrow + 32 + hi * 16);
    v8f ff0 = bfs0, ff1 = bfs1;
    ff0 = wmma16(Af00, bf0, ff0);
    ff0 = wmma16(Af01, bf1, ff0);
    ff1 = wmma16(Af10, bf0, ff1);
    ff1 = wmma16(Af11, bf1, ff1);

    // ---- ReLU + elementwise product ----
    v8f p0, p1;
    #pragma unroll
    for (int r = 0; r < 8; ++r) {
      float a0 = dd0[r] > 0.f ? dd0[r] : 0.f;
      float a1 = dd1[r] > 0.f ? dd1[r] : 0.f;
      float f0 = ff0[r] > 0.f ? ff0[r] : 0.f;
      float f1 = ff1[r] > 0.f ? ff1[r] : 0.f;
      p0[r] = a0 * f0;
      p1[r] = a1 * f1;
    }

    // ---- D-fragment -> B-fragment via cross-lane half swap ----
    v16h bq;
    #pragma unroll
    for (int r = 0; r < 8; ++r) {
      float o0 = __shfl_xor(p0[r], 16, 32);   // partner half-group, mo0
      float o1 = __shfl_xor(p1[r], 16, 32);   // partner half-group, mo1
      bq[r]     = hi ? (h16)o1    : (h16)p0[r];  // K{0..7} / K{16..23}
      bq[r + 8] = hi ? (h16)p1[r] : (h16)o0;     // K{8..15} / K{24..31}
    }

    // ---- post conv 32 -> 32 (bias as C) + ReLU, accumulate over k ----
    v8f q0 = bps0, q1 = bps1;
    q0 = wmma16(Ap0, bq, q0);
    q1 = wmma16(Ap1, bq, q1);
    #pragma unroll
    for (int r = 0; r < 8; ++r) {
      kacc0[r] += (q0[r] > 0.f ? q0[r] : 0.f);
      kacc1[r] += (q1[r] > 0.f ? q1[r] : 0.f);
    }
  }

  // ---- butterfly reduce over neighbor columns ----
  constexpr int MMAX = (K == 8) ? 8 : 16;
  #pragma unroll
  for (int m = 1; m < MMAX; m <<= 1) {
    #pragma unroll
    for (int r = 0; r < 8; ++r) {
      kacc0[r] += __shfl_xor(kacc0[r], m, 32);
      kacc1[r] += __shfl_xor(kacc1[r], m, 32);
    }
  }

  bool writer = (K == 8) ? ((c == 0) || (c == 8)) : (c == 0);
  if (writer) {
    int pt = (K == 8) ? (pb + (c >> 3)) : pb;
    h16* row = feats + (size_t)pt * 224 + coloff;
    v8h o0, o1;
    #pragma unroll
    for (int r = 0; r < 8; ++r) { o0[r] = (h16)kacc0[r]; o1[r] = (h16)kacc1[r]; }
    *(v8h*)(row + hi * 8)      = o0;   // channels  hi*8 .. hi*8+7
    *(v8h*)(row + 16 + hi * 8) = o1;   // channels 16+hi*8 ..
  }
}

// max over N per (b, ch) row of the channel-major fused_local[-1]
__global__ void maxreduce(const float* __restrict__ src, float* __restrict__ dst) {
  __shared__ float s[256];
  int row = blockIdx.x;                       // b*128 + ch
  float m = -3.4e38f;
  for (int n = threadIdx.x; n < NPTS; n += blockDim.x)
    m = fmaxf(m, src[(size_t)row * NPTS + n]);
  s[threadIdx.x] = m;
  __syncthreads();
  for (int w = blockDim.x >> 1; w > 0; w >>= 1) {
    if ((int)threadIdx.x < w) s[threadIdx.x] = fmaxf(s[threadIdx.x], s[threadIdx.x + w]);
    __syncthreads();
  }
  if (threadIdx.x == 0) dst[row] = s[0];
}

// ---------------------------------------------------------------------------
// Host orchestration
// ---------------------------------------------------------------------------
extern "C" void kernel_launch(void* const* d_in, const int* in_sizes, int n_in,
                              void* d_out, int out_size, void* d_ws, size_t ws_size,
                              hipStream_t stream) {
  (void)in_sizes; (void)n_in; (void)out_size; (void)ws_size;

  const float* pts = (const float*)d_in[0];
  const int* idxs[3] = {(const int*)d_in[1], (const int*)d_in[2], (const int*)d_in[3]};
  float* out = (float*)d_out;

  // ---- param leaf table (jax tree order: sorted dict keys) ----
  // params leaves start at d_in[4]:
  //   fusion[0]{W,b,be,g}, fusion[1]{..}, init{..},
  //   scale s: layer i {bottle,delta,feats,post}{W,b,be,g} x3, trans{W,b,be,g}
  const int NC = 42;
  int leaf[NC], cin[NC], cout[NC], bcol[NC];
  for (int c = 0; c < NC; ++c) bcol[c] = -1;
  leaf[0] = 4;  cin[0] = 2 * NFEAT; cout[0] = NFEAT;   // fusion0
  leaf[1] = 8;  cin[1] = 2 * NFEAT; cout[1] = NFEAT;   // fusion1
  leaf[2] = 12; cin[2] = 3;         cout[2] = NFEAT;   // init
  for (int s = 0; s < 3; ++s) {
    int base = 16 + 52 * s;
    int cid = 3 + 13 * s;
    for (int i = 0; i < 3; ++i) {
      int lb = base + 16 * i;
      int cl = cid + 4 * i;
      leaf[cl + 0] = lb + 0;  cin[cl + 0] = 128 + 32 * i; cout[cl + 0] = 64;  // bottle
      leaf[cl + 1] = lb + 4;  cin[cl + 1] = 3;            cout[cl + 1] = 32;  // delta
      bcol[cl + 1] = 3;       // fold delta bias into padding column 3
      leaf[cl + 2] = lb + 8;  cin[cl + 2] = 64;           cout[cl + 2] = 32;  // feats
      leaf[cl + 3] = lb + 12; cin[cl + 3] = 32;           cout[cl + 3] = 32;  // post
    }
    leaf[cid + 12] = base + 48; cin[cid + 12] = 224; cout[cid + 12] = 128;    // trans
  }

  // ---- workspace layout ----
  char* ws = (char*)d_ws;
  auto alignup = [](size_t x) { return (x + 255) & ~(size_t)255; };
  size_t off = 0;
  size_t woff[NC], boff[NC];
  for (int c = 0; c < NC; ++c) {
    int KT = (cin[c] + 31) / 32, MT = cout[c] / 16;
    woff[c] = off; off = alignup(off + (size_t)MT * KT * 512 * 2);
    boff[c] = off; off = alignup(off + (size_t)cout[c] * 4);
  }
  size_t ptsT = off;        off = alignup(off + (size_t)NP * 32 * 2);
  size_t featsS[3];
  for (int s = 0; s < 3; ++s) { featsS[s] = off; off = alignup(off + (size_t)NP * 224 * 2); }
  size_t bott = off;        off = alignup(off + (size_t)NP * 64 * 2);
  size_t fst1[3];
  for (int s = 0; s < 3; ++s) { fst1[s] = off; off = alignup(off + (size_t)NP * 128 * 2); }
  size_t fbufA = off;       off = alignup(off + (size_t)NP * 128 * 2);

  auto Wp = [&](int c) { return (h16*)(ws + woff[c]); };
  auto Bp = [&](int c) { return (float*)(ws + boff[c]); };

  // ---- 1) fold BN into weights, swizzle to A-fragment layout ----
  for (int c = 0; c < NC; ++c) {
    int KT = (cin[c] + 31) / 32, MT = cout[c] / 16;
    int total = MT * KT * 512;
    prep_conv<<<(total + 255) / 256, 256, 0, stream>>>(
        (const float*)d_in[leaf[c] + 0], (const float*)d_in[leaf[c] + 1],
        (const float*)d_in[leaf[c] + 2], (const float*)d_in[leaf[c] + 3],
        cin[c], cout[c], bcol[c], Wp(c), Bp(c));
  }

  // ---- 2) pack points ----
  pack_pts<<<NP / 256, 256, 0, stream>>>(pts, (h16*)(ws + ptsT));

  // waves -> blocks helper: PPW=4 => PTILES/4 = 256 tile groups per mo
  auto blocks = [](int MT) { return (PTILES / 4) * MT * 32 / 256; };

  // ---- 3) init conv 3->128, replicated into each scale's dense buffer ----
  for (int s = 0; s < 3; ++s) {
    gemm_cbr<1, 1, 4, false, 32, 32, 224><<<blocks(8), 256, 0, stream>>>(
        (const h16*)(ws + ptsT), (const h16*)(ws + ptsT),
        Wp(2), Bp(2), (h16*)(ws + featsS[s]), 0, nullptr);
  }

  // ---- 4) dense blocks per scale ----
  const int Ks[3] = {8, 16, 32};
  for (int s = 0; s < 3; ++s) {
    int cid = 3 + 13 * s;
    for (int i = 0; i < 3; ++i) {
      int cl = cid + 4 * i;
      const h16* fin = (const h16*)(ws + featsS[s]);
      h16* bo = (h16*)(ws + bott);
      // bottleneck: feats -> bott (64 ch), Cin = 128+32i
      if (i == 0)
        gemm_cbr<4, 4, 4, false, 224, 224, 64><<<blocks(4), 256, 0, stream>>>(
            fin, fin, Wp(cl), Bp(cl), bo, 0, nullptr);
      else if (i == 1)
        gemm_cbr<5, 5, 4, false, 224, 224, 64><<<blocks(4), 256, 0, stream>>>(
            fin, fin, Wp(cl), Bp(cl), bo, 0, nullptr);
      else
        gemm_cbr<6, 6, 4, false, 224, 224, 64><<<blocks(4), 256, 0, stream>>>(
            fin, fin, Wp(cl), Bp(cl), bo, 0, nullptr);
      // fused point3dconv -> append 32 channels at column 128+32i
      int K = Ks[s];
      int waves = (K == 8) ? NP / 2 : NP;
      int grid = waves * 32 / 256;
      if (K == 8)
        p3dconv<8><<<grid, 256, 0, stream>>>(
            pts, idxs[s], bo, Wp(cl + 1), Wp(cl + 2), Bp(cl + 2),
            Wp(cl + 3), Bp(cl + 3), (h16*)(ws + featsS[s]), 128 + 32 * i);
      else if (K == 16)
        p3dconv<16><<<grid, 256, 0, stream>>>(
            pts, idxs[s], bo, Wp(cl + 1), Wp(cl + 2), Bp(cl + 2),
            Wp(cl + 3), Bp(cl + 3), (h16*)(ws + featsS[s]), 128 + 32 * i);
      else
        p3dconv<32><<<grid, 256, 0, stream>>>(
            pts, idxs[s], bo, Wp(cl + 1), Wp(cl + 2), Bp(cl + 2),
            Wp(cl + 3), Bp(cl + 3), (h16*)(ws + featsS[s]), 128 + 32 * i);
    }
    // transition: 224 -> 128
    gemm_cbr<7, 7, 4, false, 224, 224, 128><<<blocks(8), 256, 0, stream>>>(
        (const h16*)(ws + featsS[s]), (const h16*)(ws + featsS[s]),
        Wp(cid + 12), Bp(cid + 12), (h16*)(ws + fst1[s]), 0, nullptr);
  }

  const size_t BFN = (size_t)NB * NFEAT * NPTS;   // 2097152
  float* fused0_out = out + 512;
  float* fused1_out = out + 512 + BFN;

  // ---- 5) fusion stage 0 (all stage-0 locals are init_feats, cols 0..128) ----
  gemm_cbr<8, 4, 4, false, 224, 224, 128><<<blocks(8), 256, 0, stream>>>(
      (const h16*)(ws + featsS[0]), (const h16*)(ws + featsS[1]),
      Wp(0), Bp(0), (h16*)(ws + fbufA), 0, nullptr);
  gemm_cbr<8, 4, 4, true, 128, 224, 1><<<blocks(8), 256, 0, stream>>>(
      (const h16*)(ws + fbufA), (const h16*)(ws + featsS[2]),
      Wp(1), Bp(1), nullptr, 0, fused0_out);

  // ---- 6) fusion stage 1 (trans outputs) ----
  gemm_cbr<8, 4, 4, false, 128, 128, 128><<<blocks(8), 256, 0, stream>>>(
      (const h16*)(ws + fst1[0]), (const h16*)(ws + fst1[1]),
      Wp(0), Bp(0), (h16*)(ws + fbufA), 0, nullptr);
  gemm_cbr<8, 4, 4, true, 128, 128, 1><<<blocks(8), 256, 0, stream>>>(
      (const h16*)(ws + fbufA), (const h16*)(ws + fst1[2]),
      Wp(1), Bp(1), nullptr, 0, fused1_out);

  // ---- 7) global max-pool over N ----
  maxreduce<<<NB * NFEAT, 256, 0, stream>>>(fused1_out, out);
}
